// SimilarLossCalculator_61727269978816
// MI455X (gfx1250) — matrix operand order repaired and checked by reading, not compiled, not run on hardware
//
#include <hip/hip_runtime.h>
#include <hip/hip_bf16.h>

typedef __attribute__((ext_vector_type(16))) _Float16 v16h;
typedef __attribute__((ext_vector_type(8)))  _Float16 v8h;
typedef __attribute__((ext_vector_type(8)))  float    v8f;

#define M_VIEWS 6
#define N_BATCH 2048
#define D_IN    1152
#define D_HID   128
#define D_KEY   64
#define ROWS_TOTAL (M_VIEWS * N_BATCH)   // 12288
#define KCHUNKS  (D_IN / 32)             // 36
#define NTILES   (D_HID / 16)            // 8
// final scale: mean over N (÷2048) and ÷(2*M*(M-1)) = ÷60
#define LOSS_SCALE (1.0f / (2048.0f * 60.0f))
// sqrt(1/TAU): folded into E at store time so WMMA emits sim/TAU directly
#define SQRT_INV_TAU 1.41421356237f

__global__ void sl_zero_out(float* out) {
    if (threadIdx.x == 0) out[0] = 0.0f;
}

// ---------------------------------------------------------------------------
// Stage 0: pack W (f32 [K=1152][N=128]) into f16 B-fragments in per-lane
// WMMA layout: Wp[kc][tn][lane][e] = W[(kc*32 + 16*(lane>>4) + e)*128
//                                      + tn*16 + (lane&15)]
// 288 KB total, written once, L2-resident for the whole stage-1 GEMM.
// ---------------------------------------------------------------------------
__global__ void sl_pack_w(const float* __restrict__ W, _Float16* __restrict__ Wp)
{
    const int idx  = blockIdx.x * 256 + threadIdx.x;   // 36*8*32 = 9216 frags
    const int kc   = idx >> 8;              // /256
    const int tn   = (idx >> 5) & 7;
    const int lane = idx & 31;
    const int hi   = lane >> 4;
    const int l16  = lane & 15;
    _Float16* dst = Wp + (size_t)idx * 16;
    #pragma unroll
    for (int e = 0; e < 16; ++e)
        dst[e] = (_Float16)W[(size_t)(kc * 32 + 16 * hi + e) * D_HID + tn * 16 + l16];
}

// ---------------------------------------------------------------------------
// Stage 1: x = x_all @ W + b   (12288 x 128, K=1152), f32 in, f32 out.
// One wave computes a 16x128 row stripe: A fragment converted once per
// K-chunk, reused by 8 WMMAs; B fragments come pre-packed (2 x b128 each).
// x_all (56.6 MB) is read exactly once from HBM.
// ---------------------------------------------------------------------------
__global__ void sl_gemm_xw(const float* __restrict__ xall,
                           const _Float16* __restrict__ Wp,
                           const float* __restrict__ bias,
                           float* __restrict__ x)
{
    const int wave = threadIdx.x >> 5;
    const int lane = threadIdx.x & 31;
    const int row0 = (blockIdx.x * 8 + wave) * 16;   // 768 waves total
    const int hi   = lane >> 4;                      // half-wave selector
    const int l16  = lane & 15;

    const float* arow = xall + (size_t)(row0 + l16) * D_IN;
    v8f acc[8] = {};
    for (int kc = 0; kc < KCHUNKS; ++kc) {
        const int k0 = kc * 32;
        __builtin_prefetch(arow + k0 + 128, 0, 0);   // -> global_prefetch_b8
        // A 16x32 f16, contiguous k order used here (Wp packed the same): lanes<16 K{0..15}, lanes>=16 K{16..31}
        v16h a;
        #pragma unroll
        for (int e = 0; e < 8; ++e) {
            a[e]     = (_Float16)arow[k0 + 16 * hi + e];
            a[e + 8] = (_Float16)arow[k0 + 16 * hi + 8 + e];
        }
        const _Float16* wrow = Wp + ((size_t)kc * NTILES * 32 + lane) * 16;
        #pragma unroll
        for (int tn = 0; tn < 8; ++tn) {
            v8h b0 = *(const v8h*)(wrow + (size_t)tn * 32 * 16);
            v8h b1 = *(const v8h*)(wrow + (size_t)tn * 32 * 16 + 8);
            v16h bm = __builtin_shufflevector(b0, b1,
                        0,1,2,3,4,5,6,7,8,9,10,11,12,13,14,15);
            acc[tn] = __builtin_amdgcn_wmma_f32_16x16x32_f16(false, a, false, bm,
                                                             (short)0, acc[tn],
                                                             false, false);
        }
    }
    #pragma unroll
    for (int tn = 0; tn < 8; ++tn) {
        const int ncol = tn * 16 + l16;
        const float bv = bias[ncol];
        #pragma unroll
        for (int i = 0; i < 8; ++i)   // C: VGPR i -> row i + 8*hi
            x[(size_t)(row0 + i + 8 * hi) * D_HID + ncol] = acc[tn][i] + bv;
    }
}

// ---------------------------------------------------------------------------
// Stage 2: retrieve for U and V. One wave per sample row.
//   logits = x_row @ P (64), attn = softmax(logits), E_row = attn @ P^T (128)
// E stored as f16 scaled by sqrt(1/TAU) so the pair GEMM emits sim/TAU.
// ---------------------------------------------------------------------------
__global__ void sl_retrieve(const float* __restrict__ x,
                            const float* __restrict__ U,
                            const float* __restrict__ V,
                            _Float16* __restrict__ Eu,
                            _Float16* __restrict__ Ev)
{
    __shared__ float Pl[D_HID * D_KEY];   // 32 KB
    __shared__ float attn[8][D_KEY];      // 2 KB

    const int wave = threadIdx.x >> 5;
    const int lane = threadIdx.x & 31;
    const int row  = blockIdx.x * 8 + wave;
    const float* xr = x + (size_t)row * D_HID;

    for (int which = 0; which < 2; ++which) {
        const float* P = which ? V : U;
        _Float16*    E = which ? Ev : Eu;
        __syncthreads();
        for (int i = threadIdx.x; i < D_HID * D_KEY; i += 256) Pl[i] = P[i];
        __syncthreads();

        // two logits per lane: j = lane and lane+32  (BETA == 1)
        float l0 = 0.f, l1 = 0.f;
        for (int d = 0; d < D_HID; ++d) {
            float xv = xr[d];                       // uniform addr -> cached
            l0 = fmaf(xv, Pl[d * D_KEY + lane],      l0);
            l1 = fmaf(xv, Pl[d * D_KEY + lane + 32], l1);
        }
        float mx = fmaxf(l0, l1);
        #pragma unroll
        for (int off = 16; off > 0; off >>= 1)
            mx = fmaxf(mx, __shfl_xor(mx, off, 32));
        float e0 = __expf(l0 - mx), e1 = __expf(l1 - mx);
        float s = e0 + e1;
        #pragma unroll
        for (int off = 16; off > 0; off >>= 1)
            s += __shfl_xor(s, off, 32);
        float inv = __frcp_rn(s);
        attn[wave][lane]      = e0 * inv;
        attn[wave][lane + 32] = e1 * inv;

        // E_row[d] = sum_j attn[j] * P[d][j];  d = lane + 32*i
        #pragma unroll
        for (int i = 0; i < 4; ++i) {
            int d = lane + 32 * i;
            float acc = 0.f;
            for (int j = 0; j < D_KEY; ++j)
                acc = fmaf(attn[wave][j], Pl[d * D_KEY + j], acc);
            E[(size_t)row * D_HID + d] = (_Float16)(acc * SQRT_INV_TAU);
        }
    }
}

// ---------------------------------------------------------------------------
// Stage 3: fused pair loss. grid = (8 row-blocks, 30 ordered pairs, 2 for U/V)
// Each wave owns 32 rows of E_m (two A-fragment sets resident); streams
// 16-col tiles of E_k — each B fragment feeds TWO WMMAs (8 wmma : 8 loads).
// Online logsumexp with single-exp update; diagonal element is the positive.
// sim (30 x 2048 x 2048 x 2) is never materialized.
// ---------------------------------------------------------------------------
__global__ void sl_pair_loss(const _Float16* __restrict__ Eu,
                             const _Float16* __restrict__ Ev,
                             float* __restrict__ out)
{
    const int wave = threadIdx.x >> 5;
    const int lane = threadIdx.x & 31;
    const int pair = blockIdx.y;              // 0..29
    const int m = pair / 5;
    int k = pair % 5; k += (k >= m);
    const _Float16* E  = blockIdx.z ? Ev : Eu;
    const _Float16* Em = E + (size_t)m * N_BATCH * D_HID;
    const _Float16* Ek = E + (size_t)k * N_BATCH * D_HID;

    const int hi  = lane >> 4;
    const int l16 = lane & 15;
    const int n0  = blockIdx.x * 256 + wave * 32;   // this wave's 32 rows

    // A fragments for rows n0..n0+31, K=0..127 (2 tiles x 4 chunks), resident
    v16h afr[2][4];
    #pragma unroll
    for (int t = 0; t < 2; ++t) {
        const _Float16* arow = Em + (size_t)(n0 + t * 16 + l16) * D_HID;
        #pragma unroll
        for (int kc = 0; kc < 4; ++kc) {
            const int kb = kc * 32;
            #pragma unroll
            for (int e = 0; e < 8; ++e) {
                afr[t][kc][e]     = arow[kb + 16 * hi + e];
                afr[t][kc][e + 8] = arow[kb + 16 * hi + 8 + e];
            }
        }
    }

    // per-lane online LSE state: 8 C-rows per lane per tile-set
    float mrun[2][8], srun[2][8], pos[2][8];
    #pragma unroll
    for (int t = 0; t < 2; ++t)
        #pragma unroll
        for (int i = 0; i < 8; ++i) {
            mrun[t][i] = -1e30f; srun[t][i] = 0.f; pos[t][i] = 0.f;
        }

    for (int p0 = 0; p0 < N_BATCH; p0 += 16) {
        const _Float16* brow = Ek + (size_t)(p0 + l16) * D_HID;
        __builtin_prefetch(brow + 16 * D_HID, 0, 0);   // next column tile
        v8f c0 = {}, c1 = {};
        #pragma unroll
        for (int kc = 0; kc < 4; ++kc) {
            const int kb = kc * 32 + 16 * hi;   // contiguous 16 halfs of E_k row
            v8h blo = *(const v8h*)(brow + kb);
            v8h bhi = *(const v8h*)(brow + kb + 8);
            v16h bf = __builtin_shufflevector(blo, bhi,
                        0,1,2,3,4,5,6,7,8,9,10,11,12,13,14,15);
            c0 = __builtin_amdgcn_wmma_f32_16x16x32_f16(false, afr[0][kc], false, bf,
                                                        (short)0, c0, false, false);
            c1 = __builtin_amdgcn_wmma_f32_16x16x32_f16(false, afr[1][kc], false, bf,
                                                        (short)0, c1, false, false);
        }
        const int col = p0 + l16;
        #pragma unroll
        for (int t = 0; t < 2; ++t) {
            #pragma unroll
            for (int i = 0; i < 8; ++i) {
                float s = t ? c1[i] : c0[i];          // already sim/TAU
                const int row = n0 + t * 16 + i + 8 * hi;
                if (col == row) { pos[t][i] = s; s = -__builtin_inff(); }
                // single-exp online logsumexp update
                float d  = s - mrun[t][i];
                float ex = __expf(-__builtin_fabsf(d));
                bool  up = d > 0.f;
                srun[t][i] = up ? fmaf(srun[t][i], ex, 1.f) : (srun[t][i] + ex);
                mrun[t][i] = up ? s : mrun[t][i];
            }
        }
    }

    // merge LSE state across the 16 lanes sharing each row (xor 1,2,4,8)
    float loss = 0.f;
    #pragma unroll
    for (int t = 0; t < 2; ++t) {
        #pragma unroll
        for (int i = 0; i < 8; ++i) {
            float m_ = mrun[t][i], s_ = srun[t][i], p_ = pos[t][i];
            #pragma unroll
            for (int off = 1; off < 16; off <<= 1) {
                float mo = __shfl_xor(m_, off, 32);
                float so = __shfl_xor(s_, off, 32);
                float po = __shfl_xor(p_, off, 32);
                float mn = fmaxf(m_, mo);
                s_ = s_ * __expf(m_ - mn) + so * __expf(mo - mn);
                m_ = mn;
                p_ += po;                            // exactly one lane held pos
            }
            loss += (m_ + __logf(s_)) - p_;          // lse - pos for this row
        }
    }
    if ((lane & 15) == 0)                            // lanes 0 and 16
        atomicAdd(out, loss * LOSS_SCALE);
}

// ---------------------------------------------------------------------------
extern "C" void kernel_launch(void* const* d_in, const int* in_sizes, int n_in,
                              void* d_out, int out_size, void* d_ws, size_t ws_size,
                              hipStream_t stream)
{
    const float* xall = (const float*)d_in[0];
    const float* W    = (const float*)d_in[1];
    const float* b    = (const float*)d_in[2];
    const float* U    = (const float*)d_in[3];
    const float* V    = (const float*)d_in[4];
    float* out = (float*)d_out;

    // workspace: x f32 [12288*128] | Eu f16 | Ev f16 | Wp f16  (~12.9 MB)
    float*    x  = (float*)d_ws;
    _Float16* Eu = (_Float16*)((char*)d_ws + (size_t)ROWS_TOTAL * D_HID * 4);
    _Float16* Ev = Eu + (size_t)ROWS_TOTAL * D_HID;
    _Float16* Wp = Ev + (size_t)ROWS_TOTAL * D_HID;   // 36*8*32*16 halfs = 288 KB

    sl_zero_out<<<1, 32, 0, stream>>>(out);
    sl_pack_w<<<KCHUNKS, 256, 0, stream>>>(W, Wp);           // 9216 fragments
    sl_gemm_xw<<<96, 256, 0, stream>>>(xall, Wp, b, x);      // 768 row-stripe waves
    sl_retrieve<<<ROWS_TOTAL / 8, 256, 0, stream>>>(x, U, V, Eu, Ev);
    sl_pair_loss<<<dim3(8, 30, 2), 256, 0, stream>>>(Eu, Ev, out);
}
